// Model_86535001080608
// MI455X (gfx1250) — matrix operand order repaired and checked by reading, not compiled, not run on hardware
//
#include <hip/hip_runtime.h>
#include <math.h>

// ---------------------------------------------------------------------------
// MI455X (gfx1250) RNN forward. 192-step serial chain of [256,1024]x[1024,1024]
// GEMMs on v_wmma_f32_16x16x32_bf16 (f32 accumulate). All GEMM operands kept
// bf16 so tile staging is a pure copy -> global_load_async_to_lds_b128 with
// double-buffered LDS (ASYNCcnt pipelining). Layer-0 input path algebraically
// folded: G0 = x @ (W_hidden @ Wih0) + (b_hidden @ Wih0 + bih0 + bhh0).
// ---------------------------------------------------------------------------

typedef __attribute__((ext_vector_type(16))) __bf16    bf16x16;
typedef __attribute__((ext_vector_type(8)))  float     f32x8;
typedef __attribute__((ext_vector_type(4)))  unsigned  u32x4;
typedef __attribute__((ext_vector_type(4)))  int       i32x4;

#define BM 128
#define BN 128
#define BKT 64
#define LPITCH (BKT + 8)   // LDS row pitch in halfwords: 144B = 9*16B, bank-skewed

#if __has_builtin(__builtin_amdgcn_global_load_async_to_lds_b128)
#define ASYNC_STAGE 1
#else
#define ASYNC_STAGE 0
#endif

#if ASYNC_STAGE
#if __has_builtin(__builtin_amdgcn_s_wait_asynccnt)
#define WAIT_ASYNC(n) __builtin_amdgcn_s_wait_asynccnt(n)
#else
#define WAIT_ASYNC(n) asm volatile("s_wait_asynccnt %0" ::"n"(n) : "memory")
#endif
#else
#define WAIT_ASYNC(n) ((void)0)
#endif

__device__ __forceinline__ void stage16(const unsigned short* gsrc, unsigned short* lds) {
#if ASYNC_STAGE
  __builtin_amdgcn_global_load_async_to_lds_b128(
      (__attribute__((address_space(1))) i32x4*)gsrc,
      (__attribute__((address_space(3))) i32x4*)lds, 0, 0);
#else
  *(u32x4*)lds = *(const u32x4*)gsrc;
#endif
}

__device__ __forceinline__ unsigned short f2bf(float f) {
  unsigned u = __builtin_bit_cast(unsigned, f);
  u = (u + 0x7FFFu + ((u >> 16) & 1u)) >> 16;   // round-to-nearest-even
  return (unsigned short)u;
}
__device__ __forceinline__ float bf2f(unsigned short h) {
  unsigned u = ((unsigned)h) << 16;
  return __builtin_bit_cast(float, u);
}

enum { MODE_G0 = 0, MODE_STEP0 = 1, MODE_STEP1 = 2, MODE_OUT = 3 };

struct GemmArgs {
  const unsigned short* A0;      // phase-0 A bf16, row-major, pitch lda
  const unsigned short* Bt0;     // phase-0 B transposed bf16: Bt[n][k], pitch ldb
  const unsigned short* A1;      // optional phase-1 (K-accumulated); null -> 1 phase
  const unsigned short* Bt1;
  int M, N, K, lda, ldb;
  int mode;
  const float* bias;             // f32[N]
  const unsigned short* addbf;   // bf16 additive term (G0 slice), row pitch addPitch
  int addPitch;
  float* outF;                   // MODE_OUT: remapped f32 out
  unsigned short* outBF;         // bf16 out, pitch N
  unsigned short* out2BF;        // relu bank (STEP1), pitch N; may be null
};

__global__ __launch_bounds__(256) void gemm_bf16_wmma(GemmArgs g) {
  __shared__ unsigned short As[2][BM * LPITCH];
  __shared__ unsigned short Bs[2][BN * LPITCH];

  const int tid  = threadIdx.x;
  const int lane = tid & 31;
  const int wave = tid >> 5;
  const int wm   = wave >> 2;     // 0..1 : 64-row band
  const int wn   = wave & 3;      // 0..3 : 32-col band
  const int hi   = lane >> 4;     // lane half
  const int l15  = lane & 15;

  const int m0 = blockIdx.y * BM;
  const int n0 = blockIdx.x * BN;

  f32x8 acc[4][2];
#pragma unroll
  for (int fm = 0; fm < 4; ++fm)
#pragma unroll
    for (int fn = 0; fn < 2; ++fn) acc[fm][fn] = (f32x8)0.0f;

  const int tilesPerPhase = g.K / BKT;
  const int ntiles = (g.A1 ? 2 : 1) * tilesPerPhase;

  // 4 A + 4 B async b128 per thread per tile = 8 wave-instructions/wave/tile.
  auto issue = [&](int i, int bufsel) {
    const int ph = i / tilesPerPhase;
    const int kk = (i % tilesPerPhase) * BKT;
    const unsigned short* A  = ph ? g.A1 : g.A0;
    const unsigned short* Bt = ph ? g.Bt1 : g.Bt0;
    const int row = tid >> 1, seg = tid & 1;     // 2 threads per 128B row
    {
      const unsigned short* src = A + (size_t)(m0 + row) * g.lda + kk + seg * 32;
      unsigned short* dst = &As[bufsel][row * LPITCH + seg * 32];
#pragma unroll
      for (int j = 0; j < 4; ++j) stage16(src + j * 8, dst + j * 8);
    }
    {
      int brow = n0 + row;
      if (brow >= g.N) brow = g.N - 1;           // clamp: junk cols discarded in epilogue
      const unsigned short* src = Bt + (size_t)brow * g.ldb + kk + seg * 32;
      unsigned short* dst = &Bs[bufsel][row * LPITCH + seg * 32];
#pragma unroll
      for (int j = 0; j < 4; ++j) stage16(src + j * 8, dst + j * 8);
    }
  };

  issue(0, 0);
  for (int i = 0; i < ntiles; ++i) {
    const int cur = i & 1;
    if (i + 1 < ntiles) {                 // prefetch next tile into other buffer
      issue(i + 1, cur ^ 1);
      WAIT_ASYNC(8);                      // in-order: first 8 (tile i) complete
    } else {
      WAIT_ASYNC(0);
    }
    __syncthreads();                      // all waves' tile-i copies visible

    const unsigned short* Ab = As[cur];
    const unsigned short* Bb = Bs[cur];
#pragma unroll
    for (int ks = 0; ks < BKT; ks += 32) {
      union { bf16x16 v; u32x4 q[2]; } a[4], b[2];
#pragma unroll
      for (int fm = 0; fm < 4; ++fm) {
        // A 16x32 layout: lane = M row; VGPR0-3 = K {0..7 | 8..15}, VGPR4-7 = +16
        const unsigned short* p = Ab + (wm * 64 + fm * 16 + l15) * LPITCH + ks + hi * 8;
        a[fm].q[0] = *(const u32x4*)(p);
        a[fm].q[1] = *(const u32x4*)(p + 16);
      }
#pragma unroll
      for (int fn = 0; fn < 2; ++fn) {
        // B 32x16 layout: lane = N col; VGPR0-3 = K {0..7 | 8..15}, VGPR4-7 = +16
        const unsigned short* p = Bb + (wn * 32 + fn * 16 + l15) * LPITCH + ks + hi * 8;
        b[fn].q[0] = *(const u32x4*)(p);
        b[fn].q[1] = *(const u32x4*)(p + 16);
      }
#pragma unroll
      for (int fm = 0; fm < 4; ++fm)
#pragma unroll
        for (int fn = 0; fn < 2; ++fn)
          acc[fm][fn] = __builtin_amdgcn_wmma_f32_16x16x32_bf16(
              false, a[fm].v, false, b[fn].v, (short)0, acc[fm][fn], false, false);
    }
    __syncthreads();                      // done reading buf before it is refilled
  }

  // ---- epilogue: C layout: VGPR v -> M row (hi*8+v), l15 = N col ----
  const int tileM = m0 + wm * 64;
  const int tileN = n0 + wn * 32;
#pragma unroll
  for (int fm = 0; fm < 4; ++fm) {
#pragma unroll
    for (int fn = 0; fn < 2; ++fn) {
#pragma unroll
      for (int v = 0; v < 8; ++v) {
        const int m = tileM + fm * 16 + hi * 8 + v;
        const int n = tileN + fn * 16 + l15;
        if (n >= g.N) continue;
        float val = acc[fm][fn][v];
        switch (g.mode) {
          case MODE_G0:
            g.outBF[(size_t)m * g.N + n] = f2bf(val + g.bias[n]);
            break;
          case MODE_STEP0:
            g.outBF[(size_t)m * g.N + n] =
                f2bf(tanhf(val + bf2f(g.addbf[(size_t)m * g.addPitch + n])));
            break;
          case MODE_STEP1: {
            const float h = tanhf(val + g.bias[n]);
            g.outBF[(size_t)m * g.N + n] = f2bf(h);
            if (g.out2BF) g.out2BF[(size_t)m * g.N + n] = f2bf(fmaxf(h, 0.0f));
          } break;
          case MODE_OUT: {
            const int b = m & 255, tout = m >> 8;     // GEMM row = tout*256 + b
            g.outF[(size_t)b * (48 * 64) + tout * 64 + n] = val + g.bias[n];
          } break;
        }
      }
    }
  }
}

// --------------------------- prep kernels ----------------------------------

// Wfused[k][n] = sum_j W_hidden[k][j] * Wih0[j][n]; store transposed bf16 [n][k]
__global__ void k_wfusedT(const float* __restrict__ Wh, const float* __restrict__ Wih0,
                          unsigned short* __restrict__ WfT) {
  const int idx = blockIdx.x * blockDim.x + threadIdx.x;   // 64*1024 threads
  const int k = idx >> 10, n = idx & 1023;
  float s = 0.0f;
  for (int j = 0; j < 1024; ++j) s += Wh[k * 1024 + j] * Wih0[j * 1024 + n];
  WfT[n * 64 + k] = f2bf(s);
}

// bfused[n] = b_hidden @ Wih0 + bih0 + bhh0
__global__ void k_bfused(const float* __restrict__ bh, const float* __restrict__ Wih0,
                         const float* __restrict__ bih0, const float* __restrict__ bhh0,
                         float* __restrict__ bf) {
  const int n = blockIdx.x * blockDim.x + threadIdx.x;
  if (n >= 1024) return;
  float s = bih0[n] + bhh0[n];
  for (int j = 0; j < 1024; ++j) s += bh[j] * Wih0[j * 1024 + n];
  bf[n] = s;
}

__global__ void k_addvec(const float* a, const float* b, float* c, int n) {
  const int i = blockIdx.x * blockDim.x + threadIdx.x;
  if (i < n) c[i] = a[i] + b[i];
}

__global__ void k_zero16(unsigned short* p, int n) {
  const int i = blockIdx.x * blockDim.x + threadIdx.x;
  if (i < n) p[i] = 0;
}

__global__ void k_cvt_bf16(const float* __restrict__ src, unsigned short* __restrict__ dst,
                           int n) {
  const int i = blockIdx.x * blockDim.x + threadIdx.x;
  if (i < n) dst[i] = f2bf(src[i]);
}

// dst[c][r] (bf16) = src[r][c] (f32); R x C source
__global__ void k_transpose(const float* __restrict__ src, unsigned short* __restrict__ dst,
                            int R, int C) {
  __shared__ float t[32][33];
  const int tx = threadIdx.x & 31, ty = threadIdx.x >> 5;   // 32x8
  const int c0 = blockIdx.x * 32, r0 = blockIdx.y * 32;
  for (int i = 0; i < 32; i += 8) {
    const int r = r0 + ty + i, c = c0 + tx;
    t[ty + i][tx] = (r < R && c < C) ? src[(size_t)r * C + c] : 0.0f;
  }
  __syncthreads();
  for (int i = 0; i < 32; i += 8) {
    const int c = c0 + ty + i, r = r0 + tx;
    if (c < C && r < R) dst[(size_t)c * R + r] = f2bf(t[tx][ty + i]);
  }
}

// ------------------------------ launch -------------------------------------

static inline size_t alignup(size_t x) { return (x + 255) & ~(size_t)255; }

extern "C" void kernel_launch(void* const* d_in, const int* in_sizes, int n_in,
                              void* d_out, int out_size, void* d_ws, size_t ws_size,
                              hipStream_t stream) {
  (void)in_sizes; (void)n_in; (void)out_size; (void)ws_size;
  constexpr int B = 256, S = 192, F = 64, HD = 1024, PRED = 48;
  constexpr int T0 = 143, T1 = 190;   // kept top-layer timesteps

  const float* x   = (const float*)d_in[0];
  const float* Wh  = (const float*)d_in[1];
  const float* bh  = (const float*)d_in[2];
  const float* Wih = (const float*)d_in[3];
  const float* Whh = (const float*)d_in[4];
  const float* bih = (const float*)d_in[5];
  const float* bhh = (const float*)d_in[6];
  const float* Wo  = (const float*)d_in[7];
  const float* bo  = (const float*)d_in[8];

  const float* Wih0 = Wih;                const float* Wih1 = Wih + (size_t)HD * HD;
  const float* Whh0 = Whh;                const float* Whh1 = Whh + (size_t)HD * HD;
  const float* bih0 = bih;                const float* bih1 = bih + HD;
  const float* bhh0 = bhh;                const float* bhh1 = bhh + HD;

  // workspace carve-up (all bf16 unless noted)
  char* ws = (char*)d_ws; size_t off = 0;
  auto carve = [&](size_t bytes) { char* p = ws + off; off = alignup(off + bytes); return p; };
  unsigned short* G0     = (unsigned short*)carve((size_t)B * S * HD * 2);
  unsigned short* xbf    = (unsigned short*)carve((size_t)B * S * F * 2);
  unsigned short* WfT    = (unsigned short*)carve((size_t)HD * F * 2);
  float*          bfused = (float*)carve(HD * 4);
  unsigned short* Whh0T  = (unsigned short*)carve((size_t)HD * HD * 2);
  unsigned short* Wih1T  = (unsigned short*)carve((size_t)HD * HD * 2);
  unsigned short* Whh1T  = (unsigned short*)carve((size_t)HD * HD * 2);
  unsigned short* WoT    = (unsigned short*)carve((size_t)F * HD * 2);
  float*          b1     = (float*)carve(HD * 4);
  unsigned short* h0A    = (unsigned short*)carve((size_t)B * HD * 2);
  unsigned short* h0B    = (unsigned short*)carve((size_t)B * HD * 2);
  unsigned short* h1A    = (unsigned short*)carve((size_t)B * HD * 2);
  unsigned short* h1B    = (unsigned short*)carve((size_t)B * HD * 2);
  unsigned short* Hbuf   = (unsigned short*)carve((size_t)PRED * B * HD * 2);

  // ---- weight / input prep ----
  k_transpose<<<dim3(HD / 32, HD / 32), 256, 0, stream>>>(Whh0, Whh0T, HD, HD);
  k_transpose<<<dim3(HD / 32, HD / 32), 256, 0, stream>>>(Wih1, Wih1T, HD, HD);
  k_transpose<<<dim3(HD / 32, HD / 32), 256, 0, stream>>>(Whh1, Whh1T, HD, HD);
  k_transpose<<<dim3(F / 32, HD / 32), 256, 0, stream>>>(Wo, WoT, HD, F);
  k_wfusedT<<<(F * HD) / 256, 256, 0, stream>>>(Wh, Wih0, WfT);
  k_bfused<<<4, 256, 0, stream>>>(bh, Wih0, bih0, bhh0, bfused);
  k_addvec<<<4, 256, 0, stream>>>(bih1, bhh1, b1, HD);
  k_cvt_bf16<<<(B * S * F) / 256, 256, 0, stream>>>(x, xbf, B * S * F);
  k_zero16<<<(B * HD) / 256, 256, 0, stream>>>(h0A, B * HD);
  k_zero16<<<(B * HD) / 256, 256, 0, stream>>>(h1A, B * HD);

  // ---- G0 = x @ Wfused + bfused  (all timesteps, off the serial chain) ----
  {
    GemmArgs g = {};
    g.A0 = xbf; g.Bt0 = WfT;
    g.M = B * S; g.N = HD; g.K = F; g.lda = F; g.ldb = F;
    g.mode = MODE_G0; g.bias = bfused; g.outBF = G0;
    gemm_bf16_wmma<<<dim3(HD / BN, (B * S) / BM), 256, 0, stream>>>(g);
  }

  // ---- serial recurrence ----
  unsigned short* h0[2] = {h0A, h0B};
  unsigned short* h1[2] = {h1A, h1B};
  for (int t = 0; t < S; ++t) {
    unsigned short* h0p = h0[t & 1]; unsigned short* h0c = h0[(t + 1) & 1];
    unsigned short* h1p = h1[t & 1]; unsigned short* h1c = h1[(t + 1) & 1];

    GemmArgs s0 = {};
    s0.A0 = h0p; s0.Bt0 = Whh0T;
    s0.M = B; s0.N = HD; s0.K = HD; s0.lda = HD; s0.ldb = HD;
    s0.mode = MODE_STEP0;
    s0.addbf = G0 + (size_t)t * HD; s0.addPitch = S * HD;   // G0 row = b*S + t
    s0.outBF = h0c;
    gemm_bf16_wmma<<<dim3(HD / BN, B / BM), 256, 0, stream>>>(s0);

    GemmArgs s1 = {};
    s1.A0 = h0c; s1.Bt0 = Wih1T;       // phase 0: input GEMM
    s1.A1 = h1p; s1.Bt1 = Whh1T;       // phase 1: hidden GEMM (K accumulated)
    s1.M = B; s1.N = HD; s1.K = HD; s1.lda = HD; s1.ldb = HD;
    s1.mode = MODE_STEP1; s1.bias = b1; s1.outBF = h1c;
    s1.out2BF = (t >= T0 && t <= T1) ? (Hbuf + (size_t)(t - T0) * B * HD) : nullptr;
    gemm_bf16_wmma<<<dim3(HD / BN, B / BM), 256, 0, stream>>>(s1);
  }

  // ---- out = relu(Htop) @ Wo + bo, last PRED steps ----
  {
    GemmArgs g = {};
    g.A0 = Hbuf; g.Bt0 = WoT;
    g.M = PRED * B; g.N = F; g.K = HD; g.lda = HD; g.ldb = HD;
    g.mode = MODE_OUT; g.bias = bo; g.outF = (float*)d_out;
    gemm_bf16_wmma<<<dim3((F + BN - 1) / BN, (PRED * B) / BM), 256, 0, stream>>>(g);
  }
}
